// MyLoRAAttnProcessor_59373627900237
// MI455X (gfx1250) — matrix operand order repaired and checked by reading, not compiled, not run on hardware
//
#include <hip/hip_runtime.h>
#include <hip/hip_bf16.h>
#include <stdint.h>

// ---------------------------------------------------------------------------
// Types for CDNA5 WMMA (wave32, 16x16x32 bf16 -> f32)
// ---------------------------------------------------------------------------
typedef __bf16 bf16;
typedef __attribute__((ext_vector_type(16))) __bf16 v16bf;
typedef __attribute__((ext_vector_type(8)))  __bf16 v8bf;
typedef __attribute__((ext_vector_type(8)))  float  v8f;
typedef __attribute__((ext_vector_type(4)))  float  v4f;

static __device__ __forceinline__ v16bf mk16(v8bf lo, v8bf hi) {
  return __builtin_shufflevector(lo, hi, 0,1,2,3,4,5,6,7,8,9,10,11,12,13,14,15);
}
// A-operand fragment: 16x32 bf16. Lane l (m = l&15) holds K = base + (l>>4)*16
// + s in elements s = 0..15.
// Caller passes p already offset by (l>>4)*16; we read [p, p+8) and [p+8, p+16).
static __device__ __forceinline__ v16bf ldfragA(const bf16* p) {
  v8bf lo = *(const v8bf*)p;
  v8bf hi = *(const v8bf*)(p + 8);
  return mk16(lo, hi);
}
static __device__ __forceinline__ v8f wmma_bf16(v16bf a, v16bf b, v8f c) {
  return __builtin_amdgcn_wmma_f32_16x16x32_bf16(false, a, false, b,
                                                 (short)0, c, false, false);
}

// --- CDNA5 async global->LDS copy (ASYNCcnt) via inline asm (portable) -----
typedef __attribute__((address_space(3))) __bf16 lds_bf16_t;
static __device__ __forceinline__ unsigned to_lds_off(const bf16* p) {
  return (unsigned)(unsigned long long)(const lds_bf16_t*)p;  // LDS byte offset
}
static __device__ __forceinline__ void async_ld_b128(unsigned lds_byte,
                                                     unsigned long long gaddr) {
  asm volatile("global_load_async_to_lds_b128 %0, %1, off"
               :: "v"(lds_byte), "v"(gaddr) : "memory");
}
static __device__ __forceinline__ void wait_async0() {
  asm volatile("s_wait_asynccnt 0" ::: "memory");
}

// Problem constants
#define BB   16
#define CC   1024
#define SS   4096
#define TT   77
#define TP   80      // T padded to 80 (pad rows computed as exact zeros)
#define DC   768
#define NH   16
#define DH   64

// ---------------------------------------------------------------------------
// 1) Fold LoRA into base weight:  Weff[n][k] = W[n][k] + B[n][r]*A[r][k], bf16
// ---------------------------------------------------------------------------
__global__ void fold_w(const float* __restrict__ W, const float* __restrict__ A,
                       const float* __restrict__ Bm, bf16* __restrict__ out,
                       int N, int K) {
  int idx = blockIdx.x * 256 + threadIdx.x;
  if (idx >= N * K) return;
  int n = idx / K, k = idx - n * K;
  float acc = W[idx];
#pragma unroll
  for (int r = 0; r < 16; ++r) acc += Bm[n * 16 + r] * A[r * K + k];
  out[idx] = (bf16)acc;
}

// ---------------------------------------------------------------------------
// 2a) Cast encoder states to bf16, pad T 77 -> 80 with zeros
// ---------------------------------------------------------------------------
__global__ void cast_ehs(const float* __restrict__ ehs, bf16* __restrict__ out) {
  int idx = blockIdx.x * 256 + threadIdx.x;        // over BB*TP*DC
  int d = idx % DC;
  int t = (idx / DC) % TP;
  int b = idx / (DC * TP);
  float v = (t < TT) ? ehs[((size_t)b * TT + t) * DC + d] : 0.0f;
  out[idx] = (bf16)v;
}

// ---------------------------------------------------------------------------
// 2b) Transpose+cast hidden_states [B,C,S] f32 -> xb [B*S, C] bf16 (LDS tiled)
// ---------------------------------------------------------------------------
__global__ void cast_x(const float* __restrict__ hs, bf16* __restrict__ xb) {
  __shared__ float tile[32][33];
  const int b  = blockIdx.z;
  const int c0 = blockIdx.y * 32, s0 = blockIdx.x * 32;
  const int tx = threadIdx.x, ty = threadIdx.y;    // 32 x 8
#pragma unroll
  for (int i = 0; i < 4; ++i)
    tile[ty + i * 8][tx] =
        hs[((size_t)b * CC + c0 + ty + i * 8) * SS + s0 + tx];
  __syncthreads();
#pragma unroll
  for (int i = 0; i < 4; ++i)
    xb[((size_t)b * SS + s0 + ty + i * 8) * CC + c0 + tx] =
        (bf16)tile[tx][ty + i * 8];
}

// ---------------------------------------------------------------------------
// 3/5) GEMM: C[m][n] = sum_k A[m][k] * Bw[n][k]  (bf16 A and Bw, row-major,
//      Bw in [N][K] which is exactly the WMMA B-operand per-lane layout).
//      Block = 256 thr (8 waves); block tile 256(M) x 64(N).
//      B panel staged in LDS with double-buffered async global->LDS copies
//      (ASYNCcnt), shared by all 8 waves; A streamed per-wave from global.
//      OUT_F32: add bias[n], write f32 directly in transposed [B,C,S] layout.
// ---------------------------------------------------------------------------
template <bool OUT_F32>
__global__ void __launch_bounds__(256)
gemm_wmma(const bf16* __restrict__ A, const bf16* __restrict__ Bw,
          bf16* __restrict__ Cbf, float* __restrict__ Cf32,
          const float* __restrict__ bias, int K, int N, int S) {
  __shared__ __align__(16) bf16 sB[2][64 * 64];    // 2 x 8 KB
  const int tid  = threadIdx.x;
  const int lane = tid & 31, wave = tid >> 5;
  const int half = lane >> 4, lr = lane & 15;
  const int n0 = blockIdx.x * 64;
  const int m0 = blockIdx.y * 256 + wave * 32;

  const unsigned lb[2] = { to_lds_off(&sB[0][0]), to_lds_off(&sB[1][0]) };

  // issue one 64x64 bf16 tile of Bw (rows n0..n0+63, cols kk..kk+63):
  // 512 x b128 transfers, 2 per thread (per-lane async copies).
  auto stage = [&](unsigned lbase, int kk) {
#pragma unroll
    for (int u = 0; u < 2; ++u) {
      int idx = tid + u * 256;                 // 0..511
      int n = idx >> 3, c = idx & 7;           // row, 16B chunk
      async_ld_b128(lbase + (unsigned)((n * 64 + c * 8) * 2),
                    (unsigned long long)(Bw + (size_t)(n0 + n) * K + kk + c * 8));
    }
  };

  v8f acc[2][4] = {};
  const bf16* pa0 = A + (size_t)(m0 + lr) * K + half * 16;
  const bf16* pa1 = pa0 + (size_t)16 * K;

  stage(lb[0], 0);
  const int NT = K >> 6;                       // K / 64
  for (int it = 0; it < NT; ++it) {
    const int cur = it & 1;
    const bf16* bufp = &sB[cur][0];
    wait_async0();                             // this wave's copies landed
    __syncthreads();                           // all waves' portions landed
    if (it + 1 < NT) stage(lb[cur ^ 1], (it + 1) * 64);  // overlap next tile
    const int kk = it * 64;
    if (kk + 128 < K) {
      __builtin_prefetch(pa0 + kk + 128);
      __builtin_prefetch(pa1 + kk + 128);
    }
#pragma unroll
    for (int cc = 0; cc < 2; ++cc) {           // two K=32 chunks
      v16bf a0 = ldfragA(pa0 + kk + cc * 32);
      v16bf a1 = ldfragA(pa1 + kk + cc * 32);
#pragma unroll
      for (int j = 0; j < 4; ++j) {
        v16bf bfr =
            *(const v16bf*)(bufp + (j * 16 + lr) * 64 + cc * 32 + half * 16);
        acc[0][j] = wmma_bf16(a0, bfr, acc[0][j]);
        acc[1][j] = wmma_bf16(a1, bfr, acc[1][j]);
      }
    }
    __syncthreads();                           // done reading cur buffer
  }

#pragma unroll
  for (int i = 0; i < 2; ++i)
#pragma unroll
    for (int j = 0; j < 4; ++j) {
      int n = n0 + j * 16 + lr;
      int mb = m0 + i * 16 + 8 * half;
      if constexpr (OUT_F32) {
        float bn = bias[n];
        v8f a = acc[i][j] + bn;
        int bidx = mb / S, s = mb % S;         // 8 consecutive s within one b
        float* po = Cf32 + ((size_t)bidx * N + n) * S + s;
        v4f lo = __builtin_shufflevector(a, a, 0, 1, 2, 3);
        v4f hi = __builtin_shufflevector(a, a, 4, 5, 6, 7);
        *(v4f*)po = lo;
        *(v4f*)(po + 4) = hi;
      } else {
#pragma unroll
        for (int r = 0; r < 8; ++r)
          Cbf[(size_t)(mb + r) * N + n] = (bf16)acc[i][j][r];
      }
    }
}

// ---------------------------------------------------------------------------
// 4) Attention: per (b,h,128-query block). K,Vt in LDS; QK^T WMMA; register
//    softmax with shfl_xor butterfly; P via per-wave LDS; PV WMMA.
// ---------------------------------------------------------------------------
__global__ void __launch_bounds__(256)
attn(const bf16* __restrict__ q, const bf16* __restrict__ kb,
     const bf16* __restrict__ vb, bf16* __restrict__ ob) {
  __shared__ __align__(32) bf16 kt[TP][DH];        // 80x64   (10 KB)
  __shared__ __align__(32) bf16 vt[DH][96];        // 64x96   (12 KB) V^T padded
  __shared__ __align__(32) bf16 pt[8][16][96];     // per-wave P (24 KB)

  const int tid = threadIdx.x;
  const int lane = tid & 31, wave = tid >> 5;
  const int half = lane >> 4, lr = lane & 15;
  const int b = blockIdx.z, h = blockIdx.y;
  const int qbase = blockIdx.x * 128 + wave * 16;

  // Stage K and transposed V into LDS (pad rows of kb/vb are exact zeros)
  for (int i = tid; i < TP * DH; i += 256) {
    int t = i >> 6, d = i & 63;
    size_t src = ((size_t)(b * TP + t)) * CC + h * DH + d;
    kt[t][d] = kb[src];
    vt[d][t] = vb[src];
  }
  for (int i = tid; i < DH * 16; i += 256)
    vt[i >> 4][TP + (i & 15)] = (bf16)0.0f;
  __syncthreads();

  // Q fragments: rows qbase..qbase+15, K-dim = head dims 0..63 (2 chunks)
  const bf16* pq = q + ((size_t)(b * SS + qbase + lr)) * CC + h * DH + half * 16;
  v16bf a0 = ldfragA(pq);
  v16bf a1 = ldfragA(pq + 32);

  // scores = Q @ K^T  (16 q x 80 t) : 5 N-tiles x 2 K-chunks
  v8f scacc[5];
#pragma unroll
  for (int nt = 0; nt < 5; ++nt) {
    v8f acc = {};
    v16bf b0 = *(const v16bf*)&kt[nt * 16 + lr][half * 16];
    v16bf b1 = *(const v16bf*)&kt[nt * 16 + lr][32 + half * 16];
    acc = wmma_bf16(a0, b0, acc);
    acc = wmma_bf16(a1, b1, acc);
    scacc[nt] = acc;
  }

  // softmax over t (masked t>=77), per row m = r + 8*half
  const float scale = 0.125f;                       // dh^-0.5
#pragma unroll
  for (int r = 0; r < 8; ++r) {
    int m = r + 8 * half;
    float v[5];
    float rmax = -1e30f;
#pragma unroll
    for (int nt = 0; nt < 5; ++nt) {
      int t = nt * 16 + lr;
      float s = scacc[nt][r] * scale;
      v[nt] = (t < TT) ? s : -1e30f;
      rmax = fmaxf(rmax, v[nt]);
    }
#pragma unroll
    for (int off = 8; off; off >>= 1)
      rmax = fmaxf(rmax, __shfl_xor(rmax, off, 32));
    float p[5];
    float rsum = 0.0f;
#pragma unroll
    for (int nt = 0; nt < 5; ++nt) {
      int t = nt * 16 + lr;
      p[nt] = (t < TT) ? __expf(v[nt] - rmax) : 0.0f;
      rsum += p[nt];
    }
#pragma unroll
    for (int off = 8; off; off >>= 1) rsum += __shfl_xor(rsum, off, 32);
    float rinv = 1.0f / rsum;
#pragma unroll
    for (int nt = 0; nt < 5; ++nt)
      pt[wave][m][nt * 16 + lr] = (bf16)(p[nt] * rinv);
    pt[wave][m][TP + lr] = (bf16)0.0f;              // zero pad t=80..95
  }

  // O = P @ V  (16 q x 64 d), K-dim t padded to 96: 4 N-tiles x 3 K-chunks
#pragma unroll
  for (int dt = 0; dt < 4; ++dt) {
    v8f acc = {};
#pragma unroll
    for (int c = 0; c < 3; ++c) {
      v16bf ap = ldfragA(&pt[wave][lr][c * 32 + half * 16]);
      v16bf bv = *(const v16bf*)&vt[dt * 16 + lr][c * 32 + half * 16];
      acc = wmma_bf16(ap, bv, acc);
    }
#pragma unroll
    for (int r = 0; r < 8; ++r) {
      int m = qbase + r + 8 * half;
      ob[((size_t)(b * SS + m)) * CC + h * DH + dt * 16 + lr] = (bf16)acc[r];
    }
  }
}

// ---------------------------------------------------------------------------
// Launch
// ---------------------------------------------------------------------------
extern "C" void kernel_launch(void* const* d_in, const int* in_sizes, int n_in,
                              void* d_out, int out_size, void* d_ws,
                              size_t ws_size, hipStream_t stream) {
  (void)in_sizes; (void)n_in; (void)out_size; (void)ws_size;
  const float* hs    = (const float*)d_in[0];
  const float* ehs   = (const float*)d_in[1];
  const float* Wq    = (const float*)d_in[2];
  const float* Wk    = (const float*)d_in[3];
  const float* Wv    = (const float*)d_in[4];
  const float* Wout  = (const float*)d_in[5];
  const float* b_out = (const float*)d_in[6];
  const float* Aq    = (const float*)d_in[7];
  const float* Bq    = (const float*)d_in[8];
  const float* Ak    = (const float*)d_in[9];
  const float* Bk    = (const float*)d_in[10];
  const float* Av    = (const float*)d_in[11];
  const float* Bv    = (const float*)d_in[12];
  const float* Aout  = (const float*)d_in[13];
  const float* Bout  = (const float*)d_in[14];
  float* outp = (float*)d_out;

  // Workspace carve-up (bf16 buffers)
  char* ws = (char*)d_ws;
  size_t off = 0;
  const size_t XB = (size_t)BB * SS * CC * 2;       // 128 MB
  bf16* xb   = (bf16*)(ws + off); off += XB;        // x bf16; reused as O
  bf16* qb   = (bf16*)(ws + off); off += XB;        // Q
  bf16* ehsb = (bf16*)(ws + off); off += (size_t)BB * TP * DC * 2;
  bf16* kbuf = (bf16*)(ws + off); off += (size_t)BB * TP * CC * 2;
  bf16* vbuf = (bf16*)(ws + off); off += (size_t)BB * TP * CC * 2;
  bf16* wq   = (bf16*)(ws + off); off += (size_t)CC * CC * 2;
  bf16* wk   = (bf16*)(ws + off); off += (size_t)CC * DC * 2;
  bf16* wv   = (bf16*)(ws + off); off += (size_t)CC * DC * 2;
  bf16* wo   = (bf16*)(ws + off); off += (size_t)CC * CC * 2;
  bf16* obuf = xb;                                   // alias: xb dead post-Q

  // 1) fold LoRA into weights
  fold_w<<<(CC * CC) / 256, 256, 0, stream>>>(Wq,   Aq,   Bq,   wq, CC, CC);
  fold_w<<<(CC * DC) / 256, 256, 0, stream>>>(Wk,   Ak,   Bk,   wk, CC, DC);
  fold_w<<<(CC * DC) / 256, 256, 0, stream>>>(Wv,   Av,   Bv,   wv, CC, DC);
  fold_w<<<(CC * CC) / 256, 256, 0, stream>>>(Wout, Aout, Bout, wo, CC, CC);

  // 2) casts
  cast_ehs<<<(BB * TP * DC) / 256, 256, 0, stream>>>(ehs, ehsb);
  cast_x<<<dim3(SS / 32, CC / 32, BB), dim3(32, 8), 0, stream>>>(hs, xb);

  // 3) projections (K/V: M = B*TP = 1280; Q: M = B*S = 65536)
  gemm_wmma<false><<<dim3(CC / 64, (BB * TP) / 256), 256, 0, stream>>>(
      ehsb, wk, kbuf, nullptr, nullptr, DC, CC, 0);
  gemm_wmma<false><<<dim3(CC / 64, (BB * TP) / 256), 256, 0, stream>>>(
      ehsb, wv, vbuf, nullptr, nullptr, DC, CC, 0);
  gemm_wmma<false><<<dim3(CC / 64, (BB * SS) / 256), 256, 0, stream>>>(
      xb, wq, qb, nullptr, nullptr, CC, CC, 0);

  // 4) attention
  attn<<<dim3(SS / 128, NH, BB), 256, 0, stream>>>(qb, kbuf, vbuf, obuf);

  // 5) output projection + bias, written transposed to [B,C,S] f32
  gemm_wmma<true><<<dim3(CC / 64, (BB * SS) / 256), 256, 0, stream>>>(
      obuf, wo, nullptr, outp, b_out, CC, CC, SS);
}
